// GraphSelfAttentionLayer_7438883357374
// MI455X (gfx1250) — matrix operand order repaired and checked by reading, compiled, not run on hardware
//
#include <hip/hip_runtime.h>
#include <hip/hip_bf16.h>
#include <math.h>

// Problem constants (reference: B=2, N=1024, D=128, H=64)
#define B_  2
#define N_  1024
#define D_  128
#define H_  64
#define NEG_SLOPE 0.2f

typedef __attribute__((ext_vector_type(16))) _Float16 v16h;
typedef __attribute__((ext_vector_type(8)))  _Float16 v8h;
typedef __attribute__((ext_vector_type(8)))  float    v8f;

// ---------------------------------------------------------------------------
// K0a: f32 -> f16 elementwise conversion (nodes)
// ---------------------------------------------------------------------------
__global__ void cvt_f32_f16(const float* __restrict__ src,
                            _Float16* __restrict__ dst, int n) {
  int i = blockIdx.x * blockDim.x + threadIdx.x;
  if (i < n) dst[i] = (_Float16)src[i];
}

// ---------------------------------------------------------------------------
// K0b: pack W [D x H] f32 row-major -> Wt [H x D] f16 (transposed), so the
// WMMA B-fragment (lane = N-col, K contiguous per lane) is two v8h loads.
// ---------------------------------------------------------------------------
__global__ void pack_w_t(const float* __restrict__ W, _Float16* __restrict__ Wt) {
  int i = blockIdx.x * blockDim.x + threadIdx.x;   // over D_*H_
  if (i < D_ * H_) {
    int h = i / D_, k = i % D_;
    Wt[h * D_ + k] = (_Float16)W[k * H_ + h];
  }
}

// Build the 16x32 f16 A fragment for v_wmma_f32_16x16x32_f16 from a row-major
// row pointer: lane<16 -> M=lane, K chunks [kb..kb+7] & [kb+16..kb+23];
// lane>=16 -> M=lane-16, K chunks [kb+8..15] & [kb+24..31]. Two 16B loads.
__device__ __forceinline__ v16h load_frag_rowmajor(const _Float16* rowp, int kb, int hs) {
  v8h lo = *(const v8h*)(rowp + kb + hs * 8);
  v8h hi = *(const v8h*)(rowp + kb + 16 + hs * 8);
  v16h f;
#pragma unroll
  for (int t = 0; t < 8; ++t) { f[t] = lo[t]; f[8 + t] = hi[t]; }
  return f;
}

// Build the 32x16 f16 B fragment from TRANSPOSED storage Bt[n][k]:
// lane%16 = N-col; K chunks [kb+(lane/16)*8..+7] & [kb+16+(lane/16)*8..+7].
// Two 16B loads.
__device__ __forceinline__ v16h load_bfrag_transposed(const _Float16* colp, int kb, int ks) {
  v8h lo = *(const v8h*)(colp + kb + ks * 8);
  v8h hi = *(const v8h*)(colp + kb + 16 + ks * 8);
  v16h f;
#pragma unroll
  for (int t = 0; t < 8; ++t) { f[t] = lo[t]; f[8 + t] = hi[t]; }
  return f;
}

// ---------------------------------------------------------------------------
// K1: projection GEMM  g = X @ W  via v_wmma_f32_16x16x32_f16.
//   X : [2048 x 128] f16 row-major, Wt : [64 x 128] f16 (transposed).
//   One wave = one 16x16 tile; K=128 -> 4 WMMAs, fully unrolled.
//   z==1 also emits a transposed f16 copy of g_right for K3's B operand.
// ---------------------------------------------------------------------------
__global__ __launch_bounds__(32)
void gemm_proj_wmma(const _Float16* __restrict__ Xh,
                    const _Float16* __restrict__ Wlt,
                    const _Float16* __restrict__ Wrt,
                    float* __restrict__ gl, float* __restrict__ gr,
                    _Float16* __restrict__ grt) {
  const int lane  = threadIdx.x;
  const int m0    = blockIdx.x * 16;
  const int n0    = blockIdx.y * 16;
  const int right = blockIdx.z;
  const _Float16* Wt = right ? Wrt : Wlt;

  const int l15 = lane & 15;
  const int hs  = lane >> 4;
  const _Float16* arow = Xh + (size_t)(m0 + l15) * D_;
  const _Float16* bcol = Wt + (size_t)(n0 + l15) * D_;

  v8f acc = {};
#pragma unroll
  for (int kb = 0; kb < D_; kb += 32) {
    v16h af = load_frag_rowmajor(arow, kb, hs);
    v16h bf = load_bfrag_transposed(bcol, kb, hs);
    acc = __builtin_amdgcn_wmma_f32_16x16x32_f16(false, af, false, bf,
                                                 (short)0, acc, false, false);
  }

  // C/D layout: VGPR r -> row m0 + r + 8*hs, col n0 + lane%16
  float* gout = right ? gr : gl;
#pragma unroll
  for (int r = 0; r < 8; ++r) {
    int row = m0 + r + 8 * hs;
    gout[row * H_ + n0 + l15] = acc[r];
  }
  if (right) {
#pragma unroll
    for (int r = 0; r < 8; ++r) {
      int row = m0 + r + 8 * hs;                 // 0..2047
      int bb = row >> 10, rl = row & (N_ - 1);
      grt[(size_t)bb * H_ * N_ + (size_t)(n0 + l15) * N_ + rl] = (_Float16)acc[r];
    }
  }
}

// ---------------------------------------------------------------------------
// K2: attention scores + row softmax.
//   One block per (b,i) row. g_right[b,i,:] and attn_w broadcast via LDS.
//   e[j] = sum_h lrelu(g_left[b,j,h] + g_right[b,i,h]) * w[h]
//   (attn_b is softmax-invariant and dropped.)  Output a as f16 for WMMA K3.
// ---------------------------------------------------------------------------
__global__ __launch_bounds__(256)
void attn_softmax_rows(const float* __restrict__ gl,
                       const float* __restrict__ gr,
                       const float* __restrict__ attn_w,
                       _Float16* __restrict__ a_out) {
  const int gi  = blockIdx.x;        // global row 0..2047
  const int b   = gi >> 10;
  const int i   = gi & (N_ - 1);
  const int tid = threadIdx.x;

  __shared__ float4 s_gr[H_ / 4];
  __shared__ float4 s_w[H_ / 4];
  __shared__ float  red[256];

  if (tid < H_ / 4) {
    s_gr[tid] = ((const float4*)(gr + (size_t)gi * H_))[tid];
    s_w[tid]  = ((const float4*)attn_w)[tid];
  }
  __syncthreads();

  float e[4];
#pragma unroll
  for (int jj = 0; jj < 4; ++jj) {
    int j = tid + jj * 256;
    const float4* row = (const float4*)(gl + ((size_t)b * N_ + j) * H_);
    float acc = 0.f;
#pragma unroll
    for (int q = 0; q < H_ / 4; ++q) {
      float4 g = row[q], r = s_gr[q], w = s_w[q];
      float x;
      x = g.x + r.x; acc += (x > 0.f ? x : NEG_SLOPE * x) * w.x;
      x = g.y + r.y; acc += (x > 0.f ? x : NEG_SLOPE * x) * w.y;
      x = g.z + r.z; acc += (x > 0.f ? x : NEG_SLOPE * x) * w.z;
      x = g.w + r.w; acc += (x > 0.f ? x : NEG_SLOPE * x) * w.w;
    }
    e[jj] = acc;
  }

  float lmax = fmaxf(fmaxf(e[0], e[1]), fmaxf(e[2], e[3]));
  red[tid] = lmax; __syncthreads();
  for (int s = 128; s > 0; s >>= 1) {
    if (tid < s) red[tid] = fmaxf(red[tid], red[tid + s]);
    __syncthreads();
  }
  float M = red[0]; __syncthreads();

  float p[4]; float ls = 0.f;
#pragma unroll
  for (int jj = 0; jj < 4; ++jj) { p[jj] = __expf(e[jj] - M); ls += p[jj]; }
  red[tid] = ls; __syncthreads();
  for (int s = 128; s > 0; s >>= 1) {
    if (tid < s) red[tid] += red[tid + s];
    __syncthreads();
  }
  float inv = 1.f / red[0];

  _Float16* arow = a_out + (size_t)b * N_ * N_ + (size_t)i * N_;
#pragma unroll
  for (int jj = 0; jj < 4; ++jj)
    arow[tid + jj * 256] = (_Float16)(p[jj] * inv);
}

// ---------------------------------------------------------------------------
// K3: out = a @ g_right, fused ELU.  Per batch: [1024x1024] x [1024x64].
//   One wave computes FOUR 16x16 M-tiles sharing each B fragment:
//   per k-step -> 1 B load, 4 A loads, 4 WMMAs.  K=1024 -> 32 k-steps.
//   B operand comes from the transposed f16 copy grt[b][h][n].
// ---------------------------------------------------------------------------
__global__ __launch_bounds__(32)
void attn_out_wmma(const _Float16* __restrict__ a_h,
                   const _Float16* __restrict__ grt,
                   float* __restrict__ out) {
  const int lane = threadIdx.x;
  const int m0   = blockIdx.x * 64;          // 4 M-tiles
  const int h0   = blockIdx.y * 16;
  const int b    = blockIdx.z;
  const _Float16* A  = a_h + (size_t)b * N_ * N_;
  const _Float16* Bt = grt + (size_t)b * H_ * N_;

  const int l15 = lane & 15;
  const int hs  = lane >> 4;
  const _Float16* arow0 = A + (size_t)(m0 + l15) * N_;
  const _Float16* bcol  = Bt + (size_t)(h0 + l15) * N_;

  v8f acc0 = {}, acc1 = {}, acc2 = {}, acc3 = {};
#pragma unroll 2
  for (int kb = 0; kb < N_; kb += 32) {
    v16h bf  = load_bfrag_transposed(bcol, kb, hs);
    v16h af0 = load_frag_rowmajor(arow0,               kb, hs);
    v16h af1 = load_frag_rowmajor(arow0 + 16 * N_,     kb, hs);
    v16h af2 = load_frag_rowmajor(arow0 + 32 * N_,     kb, hs);
    v16h af3 = load_frag_rowmajor(arow0 + 48 * N_,     kb, hs);
    acc0 = __builtin_amdgcn_wmma_f32_16x16x32_f16(false, af0, false, bf,
                                                  (short)0, acc0, false, false);
    acc1 = __builtin_amdgcn_wmma_f32_16x16x32_f16(false, af1, false, bf,
                                                  (short)0, acc1, false, false);
    acc2 = __builtin_amdgcn_wmma_f32_16x16x32_f16(false, af2, false, bf,
                                                  (short)0, acc2, false, false);
    acc3 = __builtin_amdgcn_wmma_f32_16x16x32_f16(false, af3, false, bf,
                                                  (short)0, acc3, false, false);
  }

  v8f accs[4] = {acc0, acc1, acc2, acc3};
#pragma unroll
  for (int t = 0; t < 4; ++t) {
#pragma unroll
    for (int r = 0; r < 8; ++r) {
      int row = m0 + 16 * t + r + 8 * hs;
      int col = h0 + l15;
      float x = accs[t][r];
      float y = x > 0.f ? x : (__expf(x) - 1.f);   // ELU(alpha=1)
      out[((size_t)b * N_ + row) * H_ + col] = y;
    }
  }
}

// ---------------------------------------------------------------------------
// Launcher
// ---------------------------------------------------------------------------
extern "C" void kernel_launch(void* const* d_in, const int* in_sizes, int n_in,
                              void* d_out, int out_size, void* d_ws, size_t ws_size,
                              hipStream_t stream) {
  const float* nodes  = (const float*)d_in[0];   // [2,1024,128]
  // d_in[1] = adj_mat (unused by the reference forward)
  const float* W_l    = (const float*)d_in[2];   // [128,64]
  const float* W_r    = (const float*)d_in[3];   // [128,64]
  const float* attn_w = (const float*)d_in[4];   // [64]
  // d_in[5] = attn_b : softmax-invariant, unused
  float* out = (float*)d_out;                    // [2,1024,64]

  // Workspace layout (all offsets 256B-aligned); total ~5.8 MB
  char* ws = (char*)d_ws;
  _Float16* X_h  = (_Float16*)(ws + 0);          // 2048*128 f16      = 524288 B
  _Float16* Wlt  = (_Float16*)(ws + 524288);     // 64*128 f16 (T)    =  16384 B
  _Float16* Wrt  = (_Float16*)(ws + 540672);     // 64*128 f16 (T)    =  16384 B
  float*    gl   = (float*)   (ws + 557056);     // 2048*64 f32       = 524288 B
  float*    gr   = (float*)   (ws + 1081344);    // 2048*64 f32       = 524288 B
  _Float16* grt  = (_Float16*)(ws + 1605632);    // [B][64][1024] f16 = 262144 B
  _Float16* a_h  = (_Float16*)(ws + 1867776);    // 2*1024*1024 f16   = 4194304 B

  // K0: f16 conversions / transposed weight packing
  cvt_f32_f16<<<(B_ * N_ * D_ + 255) / 256, 256, 0, stream>>>(nodes, X_h, B_ * N_ * D_);
  pack_w_t<<<(D_ * H_ + 255) / 256, 256, 0, stream>>>(W_l, Wlt);
  pack_w_t<<<(D_ * H_ + 255) / 256, 256, 0, stream>>>(W_r, Wrt);

  // K1: g_left / g_right projections (WMMA); 128 M-tiles x 4 N-tiles x {l,r}
  gemm_proj_wmma<<<dim3((B_ * N_) / 16, H_ / 16, 2), 32, 0, stream>>>(
      X_h, Wlt, Wrt, gl, gr, grt);

  // K2: scores + softmax -> a (f16); one block per (b,i) row
  attn_softmax_rows<<<B_ * N_, 256, 0, stream>>>(gl, gr, attn_w, a_h);

  // K3: out = a @ g_right + ELU (WMMA); 16 x 4 x 2 waves, 4 M-tiles each
  attn_out_wmma<<<dim3(N_ / 64, H_ / 16, B_), 32, 0, stream>>>(a_h, grt, out);
}
